// Model_3899830304898
// MI455X (gfx1250) — compile-verified, matrix operand order checked by reading, not run on hardware
//
#include <hip/hip_runtime.h>
#include <hip/hip_bf16.h>
#include <math.h>

// ---------------- problem constants ----------------
#define B_     16
#define T_     65536
#define H_     64
#define N_     64
#define DEPTH  4
#define TC     128                 // timesteps per chunk
#define NCHUNK (T_ / TC)           // 512
#define WFRAG  4096                // halves per pre-swizzled 64x64 weight

// ---------------- types / helpers ----------------
typedef __attribute__((ext_vector_type(16))) __bf16 v16bf;
typedef __attribute__((ext_vector_type(8)))  float  v8f;

__device__ __forceinline__ unsigned short f2bf(float f) {
    return __builtin_bit_cast(unsigned short, (__bf16)f);
}
__device__ __forceinline__ float bf2f(unsigned short s) {
    unsigned u = ((unsigned)s) << 16;
    return __builtin_bit_cast(float, u);
}

union BF16Frag {
    v16bf v;
    unsigned short s[16];
    uint4 q[2];
};

// A-fragment (16x32 bf16, row-major source with row stride 64 elements).
// ISA layout: lanes 0-15 row M=lane, halves {K k0..k0+7, k0+16..k0+23};
// lanes 16-31 same rows, k0 shifted by 8.
__device__ __forceinline__ v16bf load_a_frag(const unsigned short* rowptr, int kbase, int lane) {
    int k0 = kbase + ((lane >> 4) << 3);
    BF16Frag f;
    f.q[0] = *reinterpret_cast<const uint4*>(rowptr + k0);
    f.q[1] = *reinterpret_cast<const uint4*>(rowptr + k0 + 16);
    return f.v;
}

// B-fragment from pre-swizzled bf16 weights: 16 contiguous halves per lane.
__device__ __forceinline__ v16bf load_b_pre(const unsigned short* __restrict__ Wb,
                                            int tn, int ks, int lane) {
    const unsigned short* p = Wb + ((((tn << 1) | ks) * 32 + lane) << 4);
    BF16Frag f;
    f.q[0] = *reinterpret_cast<const uint4*>(p);
    f.q[1] = *reinterpret_cast<const uint4*>(p + 8);
    return f.v;
}

__device__ __forceinline__ v8f wmma_bf16(v16bf a, v16bf b, v8f c) {
    return __builtin_amdgcn_wmma_f32_16x16x32_bf16(false, a, false, b, (short)0, c, false, false);
}

// ---------------- prep: swizzle all weights into B-fragment layout ----------------
// wb layout: [w(0=Bm,1=Cm,2=Wlin)][layer][frag=(tn*2+ks)][lane][e0..15]
__global__ __launch_bounds__(256) void prep_weights_kernel(const float* __restrict__ Bm,
                                                           const float* __restrict__ Cm,
                                                           const float* __restrict__ Wlin,
                                                           unsigned short* __restrict__ wb) {
    int i    = blockIdx.x * 256 + threadIdx.x;   // 0 .. 3*DEPTH*WFRAG-1
    int widx = i >> 12;                          // 0..11
    int el   = i & (WFRAG - 1);
    int e    = el & 15;
    int lane = (el >> 4) & 31;
    int f    = el >> 9;                          // 0..7
    int tn   = f >> 1, ks = f & 1;
    int col  = tn * 16 + (lane & 15);
    int k0   = ks * 32 + ((lane >> 4) << 3);
    int k    = (e < 8) ? (k0 + e) : (k0 + 8 + e); // e>=8 -> k0+16+(e-8)
    int wsel = widx >> 2, l = widx & 3;
    const float* W = (wsel == 0) ? Bm : (wsel == 1) ? Cm : Wlin;
    wb[i] = f2bf(W[(l * 64 + col) * 64 + k]);
}

// ---------------- prep: per-layer recurrence constants ----------------
__global__ __launch_bounds__(64) void prep_params_kernel(const float* __restrict__ nu_log,
                                                         const float* __restrict__ gamma_log,
                                                         float* __restrict__ aArr,
                                                         float* __restrict__ gArr,
                                                         float* __restrict__ aTcArr) {
    int l = blockIdx.x, n = threadIdx.x;
    float lam = __expf(nu_log[l * 64 + n]);
    aArr[l * 64 + n]   = __expf(-lam);                 // A
    gArr[l * 64 + n]   = __expf(gamma_log[l * 64 + n]);// exp(gamma)
    aTcArr[l * 64 + n] = __expf(-lam * (float)TC);     // A^TC (closed form)
}

// ---------------- kernel 0: h = x @ Win^T  (rank-1 broadcast) ----------------
__global__ __launch_bounds__(256) void proj_in_kernel(const float* __restrict__ x,
                                                      const float* __restrict__ Win,
                                                      unsigned short* __restrict__ h) {
    long i  = (long)blockIdx.x * 256 + threadIdx.x;   // over B*T*H
    long bt = i >> 6;
    int  hh = (int)(i & 63);
    h[i] = f2bf(x[bt] * Win[hh]);
}

// ---------------- kernel 1: per-chunk Bu GEMM + local scan ----------------
__global__ __launch_bounds__(256) void lru_chunk_kernel(const unsigned short* __restrict__ u,    // [B,T,H] bf16
                                                        const unsigned short* __restrict__ WbB,  // pre-swizzled Bm
                                                        const float* __restrict__ aArr,          // [N] A
                                                        const float* __restrict__ gArr,          // [N] exp(gamma)
                                                        unsigned short* __restrict__ hloc,       // [B,T,N] bf16
                                                        float* __restrict__ carry) {             // [B,NCHUNK,N]
    __shared__ __align__(16) float Bu[TC][N_];       // 32 KB
    const int chunk = blockIdx.x, b = blockIdx.y;
    const int tid = threadIdx.x, wave = tid >> 5, lane = tid & 31;
    const long tb = (long)chunk * TC;
    const unsigned short* ubase = u + ((long)b * T_ + tb) * H_;

    const int rowa  = wave * 16 + (lane & 15);
    const int rbase = wave * 16 + ((lane >> 4) << 3);
    v16bf a0 = load_a_frag(ubase + (long)rowa * H_, 0,  lane);
    v16bf a1 = load_a_frag(ubase + (long)rowa * H_, 32, lane);
#pragma unroll
    for (int tn = 0; tn < 4; ++tn) {
        v16bf b0 = load_b_pre(WbB, tn, 0, lane);
        v16bf b1 = load_b_pre(WbB, tn, 1, lane);
        v8f c = {};
        c = wmma_bf16(a0, b0, c);
        c = wmma_bf16(a1, b1, c);
        int   n = tn * 16 + (lane & 15);
        float g = gArr[n];
#pragma unroll
        for (int r = 0; r < 8; ++r) Bu[rbase + r][n] = c[r] * g;
    }
    __syncthreads();

    // diagonal recurrence inside the chunk, zero initial state
    if (tid < N_) {
        const int n = tid;
        const float a = aArr[n];
        float h = 0.f;
        unsigned short* hp = hloc + ((long)b * T_ + tb) * N_ + n;
        for (int j = 0; j < TC; ++j) {
            h = fmaf(a, h, Bu[j][n]);
            hp[(long)j * N_] = f2bf(h);              // coalesced across the 64 threads
        }
        carry[((long)b * NCHUNK + chunk) * N_ + n] = h;
    }
}

// ---------------- kernel 2: exclusive scan over chunk carries ----------------
__global__ __launch_bounds__(64) void chunk_prefix_kernel(const float* __restrict__ carry,
                                                          const float* __restrict__ aTcArr,
                                                          float* __restrict__ prefix) {
    const int b = blockIdx.x, n = threadIdx.x;
    const float aTc = aTcArr[n];
    float p = 0.f;
    for (int c = 0; c < NCHUNK; ++c) {
        long idx = ((long)b * NCHUNK + c) * N_ + n;
        prefix[idx] = p;                              // state entering chunk c
        p = fmaf(aTc, p, carry[idx]);
    }
}

// ---------------- kernel 3: reconstruct h_prev, C-GEMM, tanh, Wlin-GEMM ----------------
__global__ __launch_bounds__(256) void lru_out_kernel(const unsigned short* __restrict__ u,     // layer input bf16
                                                      const unsigned short* __restrict__ hloc,  // local states bf16
                                                      const float* __restrict__ prefix,         // [B,NC,N]
                                                      const float* __restrict__ aArr,           // [N] A
                                                      const unsigned short* __restrict__ WbC,   // pre-swizzled Cm
                                                      const float* __restrict__ Dv,             // [H]
                                                      const unsigned short* __restrict__ WbW,   // pre-swizzled Wlin
                                                      const float* __restrict__ blin,           // [H]
                                                      unsigned short* __restrict__ out) {       // layer output bf16
    __shared__ __align__(16) unsigned short Ul[TC][64];   // u chunk
    __shared__ __align__(16) unsigned short Hl[TC][64];   // local states; reused as output stage
    __shared__ __align__(16) unsigned short HP[TC][64];   // delayed full states h_{t-1}
    __shared__ __align__(16) unsigned short Z [TC][64];   // tanh activations
    const int chunk = blockIdx.x, b = blockIdx.y;
    const int tid = threadIdx.x, wave = tid >> 5, lane = tid & 31;
    const long base = ((long)b * T_ + (long)chunk * TC) * 64;

    // cooperative, fully coalesced chunk staging (2 x 16 KB)
    {
        const uint4* gu = reinterpret_cast<const uint4*>(u + base);
        const uint4* gh = reinterpret_cast<const uint4*>(hloc + base);
        uint4* su = reinterpret_cast<uint4*>(&Ul[0][0]);
        uint4* sh = reinterpret_cast<uint4*>(&Hl[0][0]);
        for (int i = tid; i < TC * 64 / 8; i += 256) { su[i] = gu[i]; sh[i] = gh[i]; }
    }
    __syncthreads();

    // h_{t-1} = A^j * pre + hloc_{j-1}   (j = in-chunk index, pre = state before chunk)
    if (tid < N_) {
        const int n = tid;
        const float a   = aArr[n];
        const float pre = prefix[((long)b * NCHUNK + chunk) * N_ + n];
        float cur = pre, ap = 1.f;
        for (int j = 0; j < TC; ++j) {
            HP[j][n] = f2bf(cur);
            ap *= a;
            cur = fmaf(ap, pre, bf2f(Hl[j][n]));
        }
    }
    __syncthreads();

    const int rowa  = wave * 16 + (lane & 15);
    const int rbase = wave * 16 + ((lane >> 4) << 3);

    // GEMM 1: y = h_prev @ Cm^T + Dv*u ;  z = y * rsqrt(1+y^2)
    {
        v16bf a0 = load_a_frag(&HP[rowa][0], 0,  lane);
        v16bf a1 = load_a_frag(&HP[rowa][0], 32, lane);
#pragma unroll
        for (int tn = 0; tn < 4; ++tn) {
            v16bf b0 = load_b_pre(WbC, tn, 0, lane);
            v16bf b1 = load_b_pre(WbC, tn, 1, lane);
            v8f c = {};
            c = wmma_bf16(a0, b0, c);
            c = wmma_bf16(a1, b1, c);
            int   h  = tn * 16 + (lane & 15);
            float dv = Dv[h];
#pragma unroll
            for (int r = 0; r < 8; ++r) {
                float y = fmaf(dv, bf2f(Ul[rbase + r][h]), c[r]);
                float z = y * rsqrtf(fmaf(y, y, 1.f));
                Z[rbase + r][h] = f2bf(z);
            }
        }
    }
    __syncthreads();

    // GEMM 2: o = z @ Wlin^T + blin + u  (residual); stage into LDS (reuse Hl)
    {
        v16bf a0 = load_a_frag(&Z[rowa][0], 0,  lane);
        v16bf a1 = load_a_frag(&Z[rowa][0], 32, lane);
#pragma unroll
        for (int tn = 0; tn < 4; ++tn) {
            v16bf b0 = load_b_pre(WbW, tn, 0, lane);
            v16bf b1 = load_b_pre(WbW, tn, 1, lane);
            v8f c = {};
            c = wmma_bf16(a0, b0, c);
            c = wmma_bf16(a1, b1, c);
            int   h  = tn * 16 + (lane & 15);
            float bl = blin[h];
#pragma unroll
            for (int r = 0; r < 8; ++r) {
                float o = c[r] + bl + bf2f(Ul[rbase + r][h]);
                Hl[rbase + r][h] = f2bf(o);
            }
        }
    }
    __syncthreads();

    // coalesced 128-bit store of the staged output chunk
    {
        const uint4* so = reinterpret_cast<const uint4*>(&Hl[0][0]);
        uint4* go = reinterpret_cast<uint4*>(out + base);
        for (int i = tid; i < TC * 64 / 8; i += 256) go[i] = so[i];
    }
}

// ---------------- kernel 4: out = h @ Wout^T ----------------
__global__ __launch_bounds__(256) void proj_out_kernel(const unsigned short* __restrict__ h,
                                                       const float* __restrict__ Wout,
                                                       float* __restrict__ out) {
    long bt = (long)blockIdx.x * 256 + threadIdx.x;
    if (bt >= (long)B_ * T_) return;
    const uint4* q = reinterpret_cast<const uint4*>(h + bt * 64);
    float s = 0.f;
#pragma unroll
    for (int i = 0; i < 8; ++i) {
        uint4 v = q[i];
        unsigned uu[4] = {v.x, v.y, v.z, v.w};
#pragma unroll
        for (int j = 0; j < 4; ++j) {
            float lo = __builtin_bit_cast(float, uu[j] << 16);
            float hi = __builtin_bit_cast(float, uu[j] & 0xffff0000u);
            int k = i * 8 + j * 2;
            s = fmaf(lo, Wout[k], fmaf(hi, Wout[k + 1], s));
        }
    }
    out[bt] = s;
}

// ---------------- host launcher ----------------
extern "C" void kernel_launch(void* const* d_in, const int* in_sizes, int n_in,
                              void* d_out, int out_size, void* d_ws, size_t ws_size,
                              hipStream_t stream) {
    (void)in_sizes; (void)n_in; (void)out_size; (void)ws_size;
    const float* x         = (const float*)d_in[0];
    const float* Win       = (const float*)d_in[1];
    const float* nu_log    = (const float*)d_in[2];   // [D,N]
    const float* gamma_log = (const float*)d_in[3];   // [D,N]
    const float* Bm        = (const float*)d_in[4];   // [D,N,H]
    const float* Cm        = (const float*)d_in[5];   // [D,H,N]
    const float* Dv        = (const float*)d_in[6];   // [D,H]
    const float* Wlin      = (const float*)d_in[7];   // [D,H,H]
    const float* blin      = (const float*)d_in[8];   // [D,H]
    const float* Wout      = (const float*)d_in[9];   // [1,H]
    float* out = (float*)d_out;

    char* w = (char*)d_ws;
    const size_t HB = (size_t)B_ * T_ * H_ * sizeof(unsigned short);   // 128 MB
    unsigned short* hA   = (unsigned short*)(w);
    unsigned short* hB   = (unsigned short*)(w + HB);
    unsigned short* hloc = (unsigned short*)(w + 2 * HB);
    size_t off = 3 * HB;
    float* carry  = (float*)(w + off); off += (size_t)B_ * NCHUNK * N_ * sizeof(float);
    float* prefix = (float*)(w + off); off += (size_t)B_ * NCHUNK * N_ * sizeof(float);
    unsigned short* wb = (unsigned short*)(w + off); off += (size_t)3 * DEPTH * WFRAG * sizeof(unsigned short);
    float* aArr   = (float*)(w + off); off += (size_t)DEPTH * N_ * sizeof(float);
    float* gArr   = (float*)(w + off); off += (size_t)DEPTH * N_ * sizeof(float);
    float* aTcArr = (float*)(w + off); off += (size_t)DEPTH * N_ * sizeof(float);

    prep_weights_kernel<<<(3 * DEPTH * WFRAG) / 256, 256, 0, stream>>>(Bm, Cm, Wlin, wb);
    prep_params_kernel<<<DEPTH, 64, 0, stream>>>(nu_log, gamma_log, aArr, gArr, aTcArr);

    proj_in_kernel<<<(B_ * (long)T_ * H_) / 256, 256, 0, stream>>>(x, Win, hA);

    unsigned short* cur = hA;
    unsigned short* nxt = hB;
    for (int l = 0; l < DEPTH; ++l) {
        dim3 grid(NCHUNK, B_);
        const unsigned short* WbB = wb + (size_t)(0 * DEPTH + l) * WFRAG;
        const unsigned short* WbC = wb + (size_t)(1 * DEPTH + l) * WFRAG;
        const unsigned short* WbW = wb + (size_t)(2 * DEPTH + l) * WFRAG;
        lru_chunk_kernel<<<grid, 256, 0, stream>>>(cur, WbB, aArr + l * N_, gArr + l * N_,
                                                   hloc, carry);
        chunk_prefix_kernel<<<B_, N_, 0, stream>>>(carry, aTcArr + l * N_, prefix);
        lru_out_kernel<<<grid, 256, 0, stream>>>(cur, hloc, prefix, aArr + l * N_,
                                                 WbC, Dv + l * H_, WbW, blin + l * H_, nxt);
        unsigned short* t = cur; cur = nxt; nxt = t;
    }

    proj_out_kernel<<<((long)B_ * T_ + 255) / 256, 256, 0, stream>>>(cur, Wout, out);
}
